// ScanBodyBlock_27444841021627
// MI455X (gfx1250) — matrix operand order repaired and checked by reading, not compiled, not run on hardware
//
#include <hip/hip_runtime.h>
#include <cmath>

// ---------------------------------------------------------------------------
// RAFT update step for MI455X (gfx1250, wave32, WMMA + TDM).
// All convs lowered to implicit-im2col bf16 WMMA GEMMs (f32 accumulate).
// Block tile 64Mx64N, wave tile 16Mx32N (2 WMMA / K-step, shared A fragment).
// Weight tiles streamed LDS-ward by the Tensor Data Mover when available.
// ---------------------------------------------------------------------------

#define B_ 2
#define H_ 64
#define W_ 128
#define N_ (B_ * H_ * W_)   // 16384 pixels

typedef __attribute__((ext_vector_type(16))) __bf16 v16bf;
typedef __attribute__((ext_vector_type(8)))  float  v8f;
typedef __attribute__((ext_vector_type(4)))  unsigned int v4u;
typedef __attribute__((ext_vector_type(8)))  int    v8i;
typedef __attribute__((ext_vector_type(4)))  int    v4i;

#if defined(__has_builtin)
#if __has_builtin(__builtin_amdgcn_tensor_load_to_lds) && \
    __has_builtin(__builtin_amdgcn_s_wait_tensorcnt)
#define HAVE_TDM 1
#endif
#endif
#ifndef HAVE_TDM
#define HAVE_TDM 0
#endif

union Frag {
    v16bf bf;
    v4u   q[2];
};

__device__ __forceinline__ unsigned short f2bf(float f) {
    unsigned int u = __builtin_bit_cast(unsigned int, f);
    u += 0x7fffu + ((u >> 16) & 1u);          // round-to-nearest-even
    return (unsigned short)(u >> 16);
}
__device__ __forceinline__ float bf2f(unsigned short h) {
    unsigned int u = ((unsigned int)h) << 16;
    return __builtin_bit_cast(float, u);
}

// ---------------------------------------------------------------------------
// Correlation sampling: per (pixel, tap) bilinear lookup on per-pixel maps.
// Output: bf16 [N, 352] (324 real taps, 28 zero pad so K % 32 == 0).
// ---------------------------------------------------------------------------
__global__ void corr_sample_kernel(const float* __restrict__ coords1,
                                   const float* __restrict__ p0,
                                   const float* __restrict__ p1,
                                   const float* __restrict__ p2,
                                   const float* __restrict__ p3,
                                   unsigned short* __restrict__ corrB) {
    int idx = blockIdx.x * blockDim.x + threadIdx.x;
    if (idx >= N_ * 352) return;
    int n = idx / 352;
    int p = idx % 352;
    if (p >= 324) { corrB[idx] = 0; return; }

    int lvl = p / 81;
    int t   = p % 81;
    float dx = (float)(t % 9 - 4);
    float dy = (float)(t / 9 - 4);

    int Hl = H_ >> lvl;
    int Wl = W_ >> lvl;
    const float* fm;
    if      (lvl == 0) fm = p0 + (size_t)n * Hl * Wl;
    else if (lvl == 1) fm = p1 + (size_t)n * Hl * Wl;
    else if (lvl == 2) fm = p2 + (size_t)n * Hl * Wl;
    else               fm = p3 + (size_t)n * Hl * Wl;

    float scale = 1.0f / (float)(1 << lvl);
    float xs = coords1[n * 2 + 0] * scale + dx;
    float ys = coords1[n * 2 + 1] * scale + dy;

    float xf = floorf(xs), yf = floorf(ys);
    float wx = xs - xf, wy = ys - yf;
    int x0 = (int)xf, y0 = (int)yf;

    auto g = [&](int yi, int xi) -> float {
        int yc = min(max(yi, 0), Hl - 1);
        int xc = min(max(xi, 0), Wl - 1);
        float v = fm[yc * Wl + xc];
        bool valid = (yi >= 0) && (yi < Hl) && (xi >= 0) && (xi < Wl);
        return valid ? v : 0.0f;
    };
    float val = g(y0, x0)         * (1.f - wy) * (1.f - wx)
              + g(y0, x0 + 1)     * (1.f - wy) * wx
              + g(y0 + 1, x0)     * wy        * (1.f - wx)
              + g(y0 + 1, x0 + 1) * wy        * wx;
    corrB[idx] = f2bf(val);
}

// ---------------------------------------------------------------------------
// Weight repack: f32 [Kreal, CoutReal] -> bf16 TRANSPOSED [CoutPad, Kpad],
// zero padded. N-major weights make every B access contiguous in K (b128
// loads / a natural 2D TDM tile).
// ---------------------------------------------------------------------------
__global__ void wrepack_kernel(const float* __restrict__ W,
                               unsigned short* __restrict__ out,
                               int Kreal, int CoutReal, int Kpad, int CoutPad) {
    int idx = blockIdx.x * blockDim.x + threadIdx.x;
    if (idx >= Kpad * CoutPad) return;
    int c = idx / Kpad;
    int k = idx % Kpad;
    out[idx] = (k < Kreal && c < CoutReal) ? f2bf(W[(size_t)k * CoutReal + c]) : 0;
}

// ---------------------------------------------------------------------------
// Implicit-im2col WMMA GEMM:  dst[n, co] = act( sum_k A[n,k] * W[k,co] + b )
// Block = 256 threads = 8 waves. Block tile 64M x 64N; wave (mi 0..3, ni 0..1)
// computes 16M x 32N via two v_wmma_f32_16x16x32_bf16 sharing one A fragment.
// FAST (Cin % 32 == 0, Kpad == Kreal): a 32-wide K window lies inside one
// filter tap with contiguous channels -> incremental (SALU add) tap walk and
// one aligned global_load_b128 per thread for the A stage.
// B tiles (weights, n-major [CoutPad][Kpad]) are streamed into LDS by the
// Tensor Data Mover (2D tile 64 rows x 64B, LDS pad -> stride-40 rows), with
// a cooperative b128 fallback when the TDM builtin is unavailable.
// ACT: 0=none 1=relu 2=sigmoid 3=tanh
// ---------------------------------------------------------------------------
template <int ACT, bool FAST>
__global__ __launch_bounds__(256) void conv_gemm_wmma(
        const unsigned short* __restrict__ src,   // [B,H,W,Cin] bf16
        const unsigned short* __restrict__ wgt,   // [CoutPad, Kpad] bf16 (transposed)
        const float* __restrict__ bias,           // [CoutReal] f32
        unsigned short* __restrict__ dst,         // bf16, stride dstStride
        int Cin, int KH, int KW, int ph, int pw,
        int Kreal, int Kpad, int CoutPad, int CoutReal,
        int dstStride, int dstOffset) {
    constexpr int AS = 40;                    // LDS row stride (ushorts)
    __shared__ alignas(16) unsigned short sA[64 * AS];
    __shared__ alignas(16) unsigned short sB[64 * AS];

    const int tid  = threadIdx.x;
    const int lane = tid & 31;
    const int wid  = tid >> 5;
    const int mi   = wid >> 1;                // 0..3 (M subtile)
    const int ni   = wid & 1;                 // 0..1 (N subtile, 32 cols each)
    const int nb   = blockIdx.y * 64;
    const int rowBase = blockIdx.x * 64;

    // This thread's im2col row for cooperative A loads.
    const int arow = tid >> 2;                // 0..63
    const int kch  = (tid & 3) * 8;           // 0,8,16,24
    const int gRow = rowBase + arow;
    const int bImg = gRow / (H_ * W_);
    const int rem  = gRow % (H_ * W_);
    const int py   = rem / W_;
    const int px   = rem % W_;
    const size_t rowScale = (size_t)Cin;

    v8f acc0 = {};
    v8f acc1 = {};

    // Incremental filter-tap walk (uniform -> SALU adds, no div chains).
    int tky = 0, tkx = 0, ci0 = 0;

    for (int k0 = 0; k0 < Kpad; k0 += 32) {
        // ---- stage A tile ----
        if (FAST) {
            int yy = py + tky - ph;
            int xx = px + tkx - pw;
            v4u val = {};
            if (yy >= 0 && yy < H_ && xx >= 0 && xx < W_)
                val = *(const v4u*)(src +
                        (size_t)((bImg * H_ + yy) * W_ + xx) * rowScale +
                        ci0 + kch);
            *(v4u*)(sA + arow * AS + kch) = val;
            // advance tap for next K window
            ci0 += 32;
            if (ci0 >= Cin) {
                ci0 = 0;
                if (++tkx == KW) { tkx = 0; ++tky; }
            }
        } else {
            #pragma unroll
            for (int j = 0; j < 8; ++j) {
                int k = k0 + kch + j;
                unsigned short uv = 0;
                if (k < Kreal) {
                    int ci = k % Cin;
                    int s  = k / Cin;
                    int kx = s % KW;
                    int ky = s / KW;
                    int yy = py + ky - ph;
                    int xx = px + kx - pw;
                    if (yy >= 0 && yy < H_ && xx >= 0 && xx < W_)
                        uv = src[(size_t)((bImg * H_ + yy) * W_ + xx) * rowScale + ci];
                }
                sA[arow * AS + kch + j] = uv;
            }
        }

        // ---- stage B tile: 64 rows (out-channels) x 32 K, n-major ----
#if HAVE_TDM
        if (wid == 0) {
            // Tensor DMA: 2D tile, data_size=2B, tile_dim0=32 (contiguous K),
            // tile_dim1=64 rows, row stride Kpad; LDS pad 4 DWORDs every
            // 16 DWORDs -> row stride 40 ushorts (= AS).
            unsigned long long ga =
                (unsigned long long)(const void*)(wgt + (size_t)nb * (size_t)Kpad + k0);
            unsigned ldsAddr = (unsigned)(unsigned long long)(const void*)&sB[0];
            v4u g0 = { 1u,                                    // count=1 valid D#
                       ldsAddr,                               // lds_addr
                       (unsigned)ga,                          // global_addr lo
                       (unsigned)(ga >> 32) | (2u << 30) };   // addr hi | type=2
            v8i g1 = { (int)((1u << 16) |                     // data_size = 2B
                             (1u << 20) |                     // pad_enable
                             (3u << 22) |                     // pad_interval: 16 DW
                             (3u << 25)),                     // pad_amount: 4 DW
                       (int)(((unsigned)Kpad & 0xFFFFu) << 16), // tensor_dim0
                       (int)(((unsigned)CoutPad & 0xFFFFu) << 16), // tensor_dim1
                       (int)(32u << 16),                      // tile_dim0 = 32
                       (int)64,                               // tile_dim1 = 64
                       (int)Kpad,                             // tensor_dim0_stride
                       0, 0 };
            v4i z4 = { 0, 0, 0, 0 };
#if __clang_major__ >= 23
            v8i z8 = { 0, 0, 0, 0, 0, 0, 0, 0 };
            __builtin_amdgcn_tensor_load_to_lds(g0, g1, z4, z4, z8, 0);
#else
            __builtin_amdgcn_tensor_load_to_lds(g0, g1, z4, z4, 0);
#endif
            __builtin_amdgcn_s_wait_tensorcnt(0);
        }
#else
        {
            int n   = tid >> 2;               // 0..63 (out-channel row)
            v4u wv = *(const v4u*)(wgt + (size_t)(nb + n) * Kpad + k0 + kch);
            *(v4u*)(sB + n * AS + kch) = wv;
        }
        if (k0 + 32 < Kpad)                   // stream next weight tile
            __builtin_prefetch(wgt + (size_t)nb * Kpad + k0 + 32, 0, 1);
#endif
        __syncthreads();

        // ---- fragments: contiguous K order (k = 16h + s), same for A and B ----
        Frag a, b0, b1;
        {   // A: lane holds row m, K = kb..kb+15 (kb = 0 or 16)
            int kb = (lane < 16) ? 0 : 16;
            const unsigned short* ap = &sA[(mi * 16 + (lane & 15)) * AS + kb];
            a.q[0] = *(const v4u*)ap;
            a.q[1] = *(const v4u*)(ap + 8);
        }
        {   // B: lane holds col n, halves K = kb..kb+15 (kb = 0 or 16)
            int kb = (lane < 16) ? 0 : 16;
            const unsigned short* bp0 = &sB[(ni * 32 + (lane & 15)) * AS + kb];
            b0.q[0] = *(const v4u*)bp0;
            b0.q[1] = *(const v4u*)(bp0 + 8);
            const unsigned short* bp1 = bp0 + 16 * AS;
            b1.q[0] = *(const v4u*)bp1;
            b1.q[1] = *(const v4u*)(bp1 + 8);
        }
        acc0 = __builtin_amdgcn_wmma_f32_16x16x32_bf16(
                   false, a.bf, false, b0.bf, (short)0, acc0, false, false);
        acc1 = __builtin_amdgcn_wmma_f32_16x16x32_bf16(
                   false, a.bf, false, b1.bf, (short)0, acc1, false, false);
        __syncthreads();
    }

    // ---- epilogue: bias + activation, bf16 store with channel offset ----
    const int mbase = rowBase + mi * 16 + ((lane >> 4) << 3);
    const int col0  = nb + ni * 32 + (lane & 15);
    #pragma unroll
    for (int half = 0; half < 2; ++half) {
        int col = col0 + half * 16;
        if (col < CoutReal) {
            float bv = bias[col];
            const v8f& acc = half ? acc1 : acc0;
            #pragma unroll
            for (int i = 0; i < 8; ++i) {
                float v = acc[i] + bv;
                if      (ACT == 1) v = fmaxf(v, 0.0f);
                else if (ACT == 2) v = 1.0f / (1.0f + expf(-v));
                else if (ACT == 3) v = tanhf(v);
                dst[(size_t)(mbase + i) * dstStride + dstOffset + col] = f2bf(v);
            }
        }
    }
}

// ---------------------------------------------------------------------------
// Elementwise glue kernels
// ---------------------------------------------------------------------------
// hxB[:,0:128]=net, [:,128:256]=inp, [:,382:384]=flow; floB = flow (bf16)
__global__ void pack_base_kernel(const float* __restrict__ net,
                                 const float* __restrict__ inp,
                                 const float* __restrict__ coords1,
                                 const float* __restrict__ coords0,
                                 unsigned short* __restrict__ hxB,
                                 unsigned short* __restrict__ floB) {
    int idx = blockIdx.x * blockDim.x + threadIdx.x;
    if (idx >= N_ * 128) return;
    int n = idx >> 7, c = idx & 127;
    hxB[(size_t)n * 384 + c]       = f2bf(net[idx]);
    hxB[(size_t)n * 384 + 128 + c] = f2bf(inp[idx]);
    if (c < 2) {
        float fl = coords1[n * 2 + c] - coords0[n * 2 + c];
        floB[n * 2 + c] = f2bf(fl);
        hxB[(size_t)n * 384 + 382 + c] = f2bf(fl);
    }
}

// rxB[:,0:128] = r*net ; rxB[:,128:384] = hxB[:,128:384] (= x)
__global__ void build_rx_kernel(const unsigned short* __restrict__ rB,
                                const float* __restrict__ net,
                                const unsigned short* __restrict__ hxB,
                                unsigned short* __restrict__ rxB) {
    int idx = blockIdx.x * blockDim.x + threadIdx.x;
    if (idx >= N_ * 384) return;
    int n = idx / 384, c = idx % 384;
    rxB[idx] = (c < 128)
             ? f2bf(bf2f(rB[(size_t)n * 128 + c]) * net[(size_t)n * 128 + c])
             : hxB[idx];
}

// net_new = (1-z)*net + z*q  -> f32 d_out[0:N*128]  + bf16 copy for flow head
__global__ void gru_combine_kernel(const unsigned short* __restrict__ zB,
                                   const unsigned short* __restrict__ qB,
                                   const float* __restrict__ net,
                                   float* __restrict__ outNet,
                                   unsigned short* __restrict__ netNewB) {
    int idx = blockIdx.x * blockDim.x + threadIdx.x;
    if (idx >= N_ * 128) return;
    float z = bf2f(zB[idx]);
    float q = bf2f(qB[idx]);
    float nn = (1.0f - z) * net[idx] + z * q;
    outNet[idx]  = nn;
    netNewB[idx] = f2bf(nn);
}

// coords1' = coords1 + delta ; flow' = coords1' - coords0
__global__ void finalize_kernel(const unsigned short* __restrict__ deltaB,
                                const float* __restrict__ coords1,
                                const float* __restrict__ coords0,
                                float* __restrict__ out) {
    int idx = blockIdx.x * blockDim.x + threadIdx.x;
    if (idx >= N_ * 2) return;
    int n = idx >> 1, c = idx & 1;
    float d  = bf2f(deltaB[(size_t)n * 32 + c]);
    float c1 = coords1[idx] + d;
    out[(size_t)N_ * 128 + idx] = c1;               // coords1
    out[(size_t)N_ * 130 + idx] = c1 - coords0[idx];// flow
}

// ---------------------------------------------------------------------------
extern "C" void kernel_launch(void* const* d_in, const int* in_sizes, int n_in,
                              void* d_out, int out_size, void* d_ws, size_t ws_size,
                              hipStream_t stream) {
    (void)in_sizes; (void)n_in; (void)out_size; (void)ws_size;

    const float* net     = (const float*)d_in[0];
    const float* coords1 = (const float*)d_in[1];
    const float* coords0 = (const float*)d_in[2];
    const float* inp     = (const float*)d_in[3];
    const float* corr0   = (const float*)d_in[4];
    const float* corr1   = (const float*)d_in[5];
    const float* corr2   = (const float*)d_in[6];
    const float* corr3   = (const float*)d_in[7];
    const float* Wc1 = (const float*)d_in[8];   const float* bc1 = (const float*)d_in[9];
    const float* Wc2 = (const float*)d_in[10];  const float* bc2 = (const float*)d_in[11];
    const float* Wf1 = (const float*)d_in[12];  const float* bf1 = (const float*)d_in[13];
    const float* Wf2 = (const float*)d_in[14];  const float* bf2 = (const float*)d_in[15];
    const float* Wcf = (const float*)d_in[16];  const float* bcf = (const float*)d_in[17];
    const float* Wz  = (const float*)d_in[18];  const float* bz  = (const float*)d_in[19];
    const float* Wr  = (const float*)d_in[20];  const float* br  = (const float*)d_in[21];
    const float* Wq  = (const float*)d_in[22];  const float* bq  = (const float*)d_in[23];
    const float* Wfh1 = (const float*)d_in[24]; const float* bfh1 = (const float*)d_in[25];
    const float* Wfh2 = (const float*)d_in[26]; const float* bfh2 = (const float*)d_in[27];

    float* out = (float*)d_out;

    // ---- workspace carve-up (ushort units) ----
    unsigned short* ws = (unsigned short*)d_ws;
    size_t o = 0;
    auto carve = [&](size_t elems) { unsigned short* p = ws + o; o += elems; return p; };
    unsigned short* corrB   = carve((size_t)N_ * 352);
    unsigned short* cor1B   = carve((size_t)N_ * 256);
    unsigned short* catCF   = carve((size_t)N_ * 256);  // [cor2:0..192 | flo2:192..256]
    unsigned short* floB    = carve((size_t)N_ * 2);
    unsigned short* flo1B   = carve((size_t)N_ * 128);
    unsigned short* hxB     = carve((size_t)N_ * 384);  // [net|inp|m|flow]
    unsigned short* zB      = carve((size_t)N_ * 128);
    unsigned short* rB      = carve((size_t)N_ * 128);
    unsigned short* rxB     = carve((size_t)N_ * 384);
    unsigned short* qB      = carve((size_t)N_ * 128);
    unsigned short* netNewB = carve((size_t)N_ * 128);
    unsigned short* fh1B    = carve((size_t)N_ * 256);
    unsigned short* deltaB  = carve((size_t)N_ * 32);
    unsigned short* Wc1b  = carve(352 * 256);
    unsigned short* Wc2b  = carve(2304 * 192);
    unsigned short* Wf1b  = carve(128 * 128);
    unsigned short* Wf2b  = carve(1152 * 64);
    unsigned short* Wcfb  = carve(2304 * 128);
    unsigned short* Wzb   = carve(3456 * 128);
    unsigned short* Wrb   = carve(3456 * 128);
    unsigned short* Wqb   = carve(3456 * 128);
    unsigned short* Wfh1b = carve(1152 * 256);
    unsigned short* Wfh2b = carve(2304 * 64);

    auto g1 = [](size_t total) { return dim3((unsigned)((total + 255) / 256)); };

    // ---- weight repack (f32 -> transposed padded bf16 [CoutPad][Kpad]) ----
    wrepack_kernel<<<g1(352u*256), 256, 0, stream>>>(Wc1,  Wc1b,  324, 256, 352, 256);
    wrepack_kernel<<<g1(2304u*192),256, 0, stream>>>(Wc2,  Wc2b, 2304, 192, 2304, 192);
    wrepack_kernel<<<g1(128u*128), 256, 0, stream>>>(Wf1,  Wf1b,   98, 128, 128, 128);
    wrepack_kernel<<<g1(1152u*64), 256, 0, stream>>>(Wf2,  Wf2b, 1152,  64, 1152, 64);
    wrepack_kernel<<<g1(2304u*128),256, 0, stream>>>(Wcf,  Wcfb, 2304, 126, 2304, 128);
    wrepack_kernel<<<g1(3456u*128),256, 0, stream>>>(Wz,   Wzb,  3456, 128, 3456, 128);
    wrepack_kernel<<<g1(3456u*128),256, 0, stream>>>(Wr,   Wrb,  3456, 128, 3456, 128);
    wrepack_kernel<<<g1(3456u*128),256, 0, stream>>>(Wq,   Wqb,  3456, 128, 3456, 128);
    wrepack_kernel<<<g1(1152u*256),256, 0, stream>>>(Wfh1, Wfh1b,1152, 256, 1152, 256);
    wrepack_kernel<<<g1(2304u*64), 256, 0, stream>>>(Wfh2, Wfh2b,2304,   2, 2304, 64);

    // ---- correlation features + base packing ----
    corr_sample_kernel<<<g1((size_t)N_ * 352), 256, 0, stream>>>(
        coords1, corr0, corr1, corr2, corr3, corrB);
    pack_base_kernel<<<g1((size_t)N_ * 128), 256, 0, stream>>>(
        net, inp, coords1, coords0, hxB, floB);

    auto gemm = [](int CoutPad) { return dim3(N_ / 64, CoutPad / 64); };

    // motion encoder
    conv_gemm_wmma<1, true><<<gemm(256), 256, 0, stream>>>(   // 1x1 324->256
        corrB, Wc1b, bc1, cor1B, 352, 1, 1, 0, 0, 352, 352, 256, 256, 256, 0);
    conv_gemm_wmma<1, true><<<gemm(192), 256, 0, stream>>>(   // 3x3 256->192
        cor1B, Wc2b, bc2, catCF, 256, 3, 3, 1, 1, 2304, 2304, 192, 192, 256, 0);
    conv_gemm_wmma<1, false><<<gemm(128), 256, 0, stream>>>(  // 7x7 2->128 (generic)
        floB, Wf1b, bf1, flo1B, 2, 7, 7, 3, 3, 98, 128, 128, 128, 128, 0);
    conv_gemm_wmma<1, true><<<gemm(64), 256, 0, stream>>>(    // 3x3 128->64
        flo1B, Wf2b, bf2, catCF, 128, 3, 3, 1, 1, 1152, 1152, 64, 64, 256, 192);
    conv_gemm_wmma<1, true><<<gemm(128), 256, 0, stream>>>(   // 3x3 256->126 (m)
        catCF, Wcfb, bcf, hxB, 256, 3, 3, 1, 1, 2304, 2304, 128, 126, 384, 256);

    // ConvGRU
    conv_gemm_wmma<2, true><<<gemm(128), 256, 0, stream>>>(   // z = sigmoid
        hxB, Wzb, bz, zB, 384, 3, 3, 1, 1, 3456, 3456, 128, 128, 128, 0);
    conv_gemm_wmma<2, true><<<gemm(128), 256, 0, stream>>>(   // r = sigmoid
        hxB, Wrb, br, rB, 384, 3, 3, 1, 1, 3456, 3456, 128, 128, 128, 0);
    build_rx_kernel<<<g1((size_t)N_ * 384), 256, 0, stream>>>(rB, net, hxB, rxB);
    conv_gemm_wmma<3, true><<<gemm(128), 256, 0, stream>>>(   // q = tanh
        rxB, Wqb, bq, qB, 384, 3, 3, 1, 1, 3456, 3456, 128, 128, 128, 0);
    gru_combine_kernel<<<g1((size_t)N_ * 128), 256, 0, stream>>>(
        zB, qB, net, out, netNewB);

    // flow head
    conv_gemm_wmma<1, true><<<gemm(256), 256, 0, stream>>>(   // 3x3 128->256
        netNewB, Wfh1b, bfh1, fh1B, 128, 3, 3, 1, 1, 1152, 1152, 256, 256, 256, 0);
    conv_gemm_wmma<0, true><<<gemm(64), 256, 0, stream>>>(    // 3x3 256->2
        fh1B, Wfh2b, bfh2, deltaB, 256, 3, 3, 1, 1, 2304, 2304, 64, 2, 32, 0);

    finalize_kernel<<<g1((size_t)N_ * 2), 256, 0, stream>>>(
        deltaB, coords1, coords0, out);
}
